// ConvBnA_int_10814727652051
// MI455X (gfx1250) — 8-bit matrix operand order repaired and checked by reading, compiled, not run on hardware
//
#include <hip/hip_runtime.h>
#include <stdint.h>

typedef __attribute__((ext_vector_type(8))) int v8i;

#define BB   32
#define CIN  128
#define COUT 256
#define HH   56
#define WW   56
#define HW   (HH * WW)
#define NPIX (BB * HW)
#define KTOT (CIN * 9)
#define NCH  18
#define LDSS 80

__device__ __forceinline__ void async_copy16(uint32_t lds_addr, const void* gaddr) {
  asm volatile("global_load_async_to_lds_b128 %0, %1, off"
               :: "v"(lds_addr), "v"(gaddr)
               : "memory");
}
__device__ __forceinline__ void wait_async0() {
  asm volatile("s_wait_asynccnt 0" ::: "memory");
}

__global__ void quant_x_kernel(const float* __restrict__ x, uint32_t* __restrict__ xq) {
  int od = blockIdx.x * blockDim.x + threadIdx.x;
  if (od >= NPIX * CIN / 4) return;
  int pix = od >> 5;
  int c4  = (od & 31) << 2;
  int b   = pix / HW;
  int phw = pix - b * HW;
  const float* src = x + (size_t)(b * CIN + c4) * HW + phw;
  uint32_t v = 0;
  v |= ((uint32_t)(int)__builtin_nontemporal_load(src)          & 0xFFu);
  v |= ((uint32_t)(int)__builtin_nontemporal_load(src + HW)     & 0xFFu) << 8;
  v |= ((uint32_t)(int)__builtin_nontemporal_load(src + 2 * HW) & 0xFFu) << 16;
  v |= ((uint32_t)(int)__builtin_nontemporal_load(src + 3 * HW) & 0xFFu) << 24;
  xq[od] = v;
}

__global__ void pack_w_kernel(const float* __restrict__ w, uint32_t* __restrict__ wp) {
  int od = blockIdx.x * blockDim.x + threadIdx.x;
  if (od >= COUT * KTOT / 4) return;
  int co  = od / 288;
  int rem = od - co * 288;
  int t   = rem >> 4;
  int j   = rem & 15;
  int p    = (j >> 1) & 3;
  int lo   = j & 1;
  int half = j >> 3;
  int l    = lo + (p & 1) * 2 + (p >> 1) * 8 + half * 4;
  int khw  = t >> 1;
  int ky   = khw / 3, kx = khw - ky * 3;
  int cinB = (t & 1) * 64 + l * 4;
  uint32_t v = 0;
  #pragma unroll
  for (int e = 0; e < 4; ++e) {
    int cin = cinB + e;
    float f = w[(((size_t)co * CIN + cin) * 3 + ky) * 3 + kx];
    v |= ((uint32_t)(int)f & 0xFFu) << (8 * e);
  }
  wp[od] = v;
}

__global__ __launch_bounds__(256)
void conv_wmma_kernel(const uint8_t* __restrict__ xq,
                      const uint8_t* __restrict__ wp,
                      const float*   __restrict__ tb,
                      const int*     __restrict__ nsh,
                      const int*     __restrict__ amin,
                      const int*     __restrict__ amax,
                      float*         __restrict__ out) {
  __shared__ __align__(16) uint8_t sA[2][128 * LDSS];
  __shared__ __align__(16) uint8_t sB[2][128 * LDSS];

  const int tid   = threadIdx.x;
  const int lane  = tid & 31;
  const int wave  = tid >> 5;
  const int waveM = wave & 1;
  const int waveN = wave >> 1;
  const int m16   = lane & 15;
  const int h     = lane >> 4;

  const int n0  = blockIdx.x * 128;
  const int co0 = blockIdx.y * 128;

  const int frow  = tid >> 1;
  const int fhalf = tid & 1;
  const int fn    = n0 + frow;
  const int fb    = fn / HW;
  const int fphw  = fn - fb * HW;
  const int fy    = fphw / WW;
  const int fx    = fphw - fy * WW;
  const uint8_t* aSrcRow = wp + (size_t)(co0 + frow) * KTOT + fhalf * 32;

  auto issue_fill = [&](int t, int buf) {
    uint32_t aLds = (uint32_t)(uintptr_t)&sA[buf][frow * LDSS + fhalf * 32];
    const uint8_t* aG = aSrcRow + t * 64;
    async_copy16(aLds,      aG);
    async_copy16(aLds + 16, aG + 16);
    uint32_t bLds = (uint32_t)(uintptr_t)&sB[buf][frow * LDSS + fhalf * 32];
    int khw = t >> 1;
    int ky = khw / 3, kx = khw - ky * 3;
    int iy = fy + ky - 1, ix = fx + kx - 1;
    if ((unsigned)iy < (unsigned)HH && (unsigned)ix < (unsigned)WW) {
      const uint8_t* bG = xq + (size_t)((fb * HH + iy) * WW + ix) * CIN
                          + (t & 1) * 64 + fhalf * 32;
      async_copy16(bLds,      bG);
      async_copy16(bLds + 16, bG + 16);
    } else {
      const uint4 z = make_uint4(0u, 0u, 0u, 0u);
      uint4* bd = (uint4*)(&sB[buf][frow * LDSS + fhalf * 32]);
      bd[0] = z; bd[1] = z;
    }
  };

  v8i acc[4][2];
  const v8i vz = {0, 0, 0, 0, 0, 0, 0, 0};
  #pragma unroll
  for (int mi = 0; mi < 4; ++mi)
    #pragma unroll
    for (int ni = 0; ni < 2; ++ni) acc[mi][ni] = vz;

  issue_fill(0, 0);
  wait_async0();
  __syncthreads();

  union V8 { uint4 q[2]; v8i v; };

  for (int t = 0; t < NCH; ++t) {
    const int cur = t & 1;
    if (t + 1 < NCH) issue_fill(t + 1, cur ^ 1);

    V8 af[4];
    #pragma unroll
    for (int mi = 0; mi < 4; ++mi) {
      const uint8_t* p = &sA[cur][(waveM * 64 + mi * 16 + m16) * LDSS + h * 32];
      af[mi].q[0] = *(const uint4*)(p);
      af[mi].q[1] = *(const uint4*)(p + 16);
    }
    V8 bf[2];
    #pragma unroll
    for (int ni = 0; ni < 2; ++ni) {
      const uint8_t* p = &sB[cur][(waveN * 32 + ni * 16 + m16) * LDSS + h * 16];
      bf[ni].q[0] = *(const uint4*)(p);
      bf[ni].q[1] = *(const uint4*)(p + 32);
    }

    #pragma unroll
    for (int mi = 0; mi < 4; ++mi)
      #pragma unroll
      for (int ni = 0; ni < 2; ++ni)
        acc[mi][ni] = __builtin_amdgcn_wmma_i32_16x16x64_iu8(
             true,  af[mi].v,
             false, bf[ni].v,
            acc[mi][ni],  false,  false);

    wait_async0();
    __syncthreads();
  }

  #pragma unroll
  for (int mi = 0; mi < 4; ++mi) {
    const int coBase = co0 + waveM * 64 + mi * 16 + h * 8;
    int bias[8], sh[8], lo[8], hi[8];
    #pragma unroll
    for (int j = 0; j < 8; ++j) {
      const int co = coBase + j;
      bias[j] = (int)tb[co];
      sh[j]   = -nsh[co];
      lo[j]   = amin[co];
      hi[j]   = amax[co];
    }
    #pragma unroll
    for (int ni = 0; ni < 2; ++ni) {
      const int npix = n0 + waveN * 32 + ni * 16 + m16;
      const int ob   = npix / HW;
      const int ophw = npix - ob * HW;
      #pragma unroll
      for (int j = 0; j < 8; ++j) {
        int v = acc[mi][ni][j] + bias[j];
        v >>= sh[j];
        v = (v < lo[j]) ? lo[j] : v;
        v = (v > hi[j]) ? hi[j] : v;
        __builtin_nontemporal_store(
            (float)v, out + ((size_t)(ob * COUT + coBase + j)) * HW + ophw);
      }
    }
  }
}

extern "C" void kernel_launch(void* const* d_in, const int* in_sizes, int n_in,
                              void* d_out, int out_size, void* d_ws, size_t ws_size,
                              hipStream_t stream) {
  const float* x    = (const float*)d_in[0];
  const float* w    = (const float*)d_in[1];
  const float* tb   = (const float*)d_in[2];
  const int*   nsh  = (const int*)d_in[3];
  const int*   amin = (const int*)d_in[4];
  const int*   amax = (const int*)d_in[5];
  float* out = (float*)d_out;

  uint8_t* xq  = (uint8_t*)d_ws;
  uint8_t* wpk = xq + (size_t)NPIX * CIN;

  {
    int nd = NPIX * CIN / 4;
    quant_x_kernel<<<(nd + 255) / 256, 256, 0, stream>>>(x, (uint32_t*)xq);
  }
  {
    int nd = COUT * KTOT / 4;
    pack_w_kernel<<<(nd + 255) / 256, 256, 0, stream>>>(w, (uint32_t*)wpk);
  }
  dim3 grid(NPIX / 128, COUT / 128);
  conv_wmma_kernel<<<grid, 256, 0, stream>>>(xq, wpk, tb, nsh, amin, amax, out);
}
